// GCN_4174708212101
// MI455X (gfx1250) — matrix operand order repaired and checked by reading, not compiled, not run on hardware
//
#include <hip/hip_runtime.h>

#define N_NODES 100000
#define N_EDGES 1600000
#define HDIM    128
#define NGRAPH  512

typedef __attribute__((ext_vector_type(16))) __bf16 v16bf;
typedef __attribute__((ext_vector_type(8)))  float  v8f;
typedef __attribute__((ext_vector_type(4)))  unsigned int u32x4;
typedef __attribute__((ext_vector_type(8)))  int i32x8;
typedef __attribute__((ext_vector_type(4)))  int i32x4;

union Frag16 { v16bf v; unsigned int u[8]; };

__device__ __forceinline__ unsigned int pack_bf16(float a, float b) {
#if defined(__gfx1250__) && __has_builtin(__builtin_amdgcn_cvt_pk_bf16_f32)
    typedef __attribute__((ext_vector_type(2))) __bf16 v2bf;
    union { v2bf v; unsigned int u; } r;
    r.v = __builtin_amdgcn_cvt_pk_bf16_f32(a, b);
    return r.u;
#else
    unsigned int ua = __float_as_uint(a);
    unsigned int ub = __float_as_uint(b);
    ua = (ua + 0x7FFFu + ((ua >> 16) & 1u)) >> 16;   // round-to-nearest-even
    ub = (ub + 0x7FFFu + ((ub >> 16) & 1u)) >> 16;
    return (ua & 0xFFFFu) | (ub << 16);
#endif
}

// ---------------- degree / dinv ----------------
__global__ void deg_init(float* deg) {
    int i = blockIdx.x * 256 + threadIdx.x;
    if (i < N_NODES) deg[i] = 1.0f;            // self-loop counts in degree
}
__global__ void deg_acc(float* deg, const int* __restrict__ dst) {
    int e = blockIdx.x * 256 + threadIdx.x;
    if (e < N_EDGES) atomicAdd(&deg[dst[e]], 1.0f);
}
__global__ void deg_rsqrt(float* deg) {
    int i = blockIdx.x * 256 + threadIdx.x;
    if (i < N_NODES) deg[i] = rsqrtf(deg[i]);
}

// ---------------- one-time input conversion: f32 rows -> bf16 rows ---------
__global__ void xprep(const float* __restrict__ X, unsigned int* __restrict__ Xbf) {
    long i = (long)blockIdx.x * 256 + threadIdx.x;    // dword index over N*64
    if (i < (long)N_NODES * (HDIM / 2)) {
        float2 f = *(const float2*)(X + 2 * i);
        Xbf[i] = pack_bf16(f.x, f.y);
    }
}

// ---------------- per-layer weight pre-swizzle: f32 [k][n] -> bf16 fragments
// wsw[ ((t*4 + c)*32 + lane)*8 + v ] = pack( W[k0][n], W[k0+1][n] )
// with n = t*16 + (lane&15), k0 = c*32 + (lane>=16 ? 8 : 0) + (v<4 ? 2v : 16 + 2(v-4)).
__global__ void wprep(const float* __restrict__ W, unsigned int* __restrict__ wsw) {
    int d    = blockIdx.x * 256 + threadIdx.x;     // 0..8191
    int v    = d & 7;
    int lane = (d >> 3) & 31;
    int c    = (d >> 8) & 3;
    int t    = d >> 10;
    int n    = t * 16 + (lane & 15);
    int k0   = c * 32 + ((lane >> 4) << 3) + ((v & 3) << 1) + ((v >> 2) << 4);
    wsw[d] = pack_bf16(W[k0 * HDIM + n], W[(k0 + 1) * HDIM + n]);
}

__device__ __forceinline__ void loadB(Frag16& f, const unsigned int* wsh,
                                      int lane, int c, int t) {
    const uint4* bp = (const uint4*)(wsh + ((t * 4 + c) * 32 + lane) * 8);
    uint4 q0 = bp[0], q1 = bp[1];
    f.u[0] = q0.x; f.u[1] = q0.y; f.u[2] = q0.z; f.u[3] = q0.w;
    f.u[4] = q1.x; f.u[5] = q1.y; f.u[6] = q1.z; f.u[7] = q1.w;
}

// ---------------- GEMM: Y[nrows x 128] = X[nrows x 128] * W[128 x 128] -----
// A is pre-converted bf16 (row-major, 64 dwords/row): fragment = two b128
// loads, no conversion VALU. W is pre-swizzled bf16, DMA'd to LDS by the TDM.
// B fragments are register double-buffered: the loads for tile t+1 issue
// before the WMMA for tile t, keeping one fragment in flight behind each MMA.
__global__ __launch_bounds__(256) void gemm_xw(const unsigned int* __restrict__ Xbf,
                                               const unsigned int* __restrict__ Wsw,
                                               float* __restrict__ Y, int nrows) {
    __shared__ __align__(16) unsigned int wsh[8192];   // 32 KB bf16 swizzled W

    const int tid = threadIdx.x;

#if defined(__gfx1250__) && __has_builtin(__builtin_amdgcn_tensor_load_to_lds)
    if (tid < 32) {
        // Tensor DMA descriptor (D#): 1-D tile of 8192 dwords.
        unsigned long long ga = (unsigned long long)Wsw;
        unsigned lds = (unsigned)(unsigned long long)(void*)wsh;  // low 32 bits = LDS offset
        u32x4 g0;
        g0[0] = 1u;                                              // count=1, user mode
        g0[1] = lds;                                             // lds_addr
        g0[2] = (unsigned)ga;                                    // global_addr[31:0]
        g0[3] = (unsigned)((ga >> 32) & 0x01FFFFFFu) | (2u << 30); // addr[56:32] | type=2
        i32x8 g1;
        g1[0] = (int)(2u << 16);                   // data_size=4B, mask=0, no flags
        g1[1] = (int)((8192u & 0xFFFFu) << 16);    // tensor_dim0[15:0]
        g1[2] = (int)((8192u >> 16) | (1u << 16)); // tensor_dim0 hi | tensor_dim1=1
        g1[3] = (int)((8192u & 0xFFFFu) << 16);    // tile_dim0=8192
        g1[4] = 1;                                 // tile_dim1=1, tile_dim2=0
        g1[5] = 8192;                              // tensor_dim0_stride lo32
        g1[6] = 0;
        g1[7] = 0;
        i32x4 gz = {0, 0, 0, 0};
#if __has_include(<hip/amd_detail/amd_gfx1250_TDM.h>)
        i32x8 gz8 = {0, 0, 0, 0, 0, 0, 0, 0};
        __builtin_amdgcn_tensor_load_to_lds(g0, g1, gz, gz, gz8, 0);
#else
        __builtin_amdgcn_tensor_load_to_lds(g0, g1, gz, gz, 0);
#endif
        __builtin_amdgcn_s_wait_tensorcnt(0);
    }
#else
#pragma unroll
    for (int it = 0; it < 32; ++it) wsh[tid + it * 256] = Wsw[tid + it * 256];
#endif
    __syncthreads();

    const int wave = tid >> 5;
    const int lane = tid & 31;
    int row  = blockIdx.x * 128 + wave * 16 + (lane & 15);
    int rcl  = row < nrows ? row : (nrows - 1);
    const unsigned int* xr = Xbf + (long)rcl * (HDIM / 2);  // bf16 row, 64 dwords
    const int koff2 = (lane >> 4) << 2;                     // 0 or 4 dwords (K offset 0/8)

    v8f acc[8] = {};

    Frag16 bcur, bnext;
    loadB(bcur, wsh, lane, 0, 0);

#pragma unroll
    for (int c = 0; c < 4; ++c) {
        // A fragment: lane<16 -> K = c*32 + {0..7, 16..23}; lane>=16 -> +8.
        const unsigned int* xa = xr + c * 16 + koff2;
        uint4 p0 = *(const uint4*)(xa);       // K = c*32 + koff .. +7
        uint4 p1 = *(const uint4*)(xa + 8);   // K = c*32 + koff+16 .. +23
        Frag16 af;
        af.u[0] = p0.x; af.u[1] = p0.y; af.u[2] = p0.z; af.u[3] = p0.w;
        af.u[4] = p1.x; af.u[5] = p1.y; af.u[6] = p1.z; af.u[7] = p1.w;
#pragma unroll
        for (int t = 0; t < 8; ++t) {
            if (!(c == 3 && t == 7)) {
                int nc = (t == 7) ? c + 1 : c;
                int nt = (t == 7) ? 0 : t + 1;
                loadB(bnext, wsh, lane, nc, nt);
            }
            acc[t] = __builtin_amdgcn_wmma_f32_16x16x32_bf16(
                false, af.v, false, bcur.v, (short)0, acc[t], false, false);
            bcur = bnext;   // SSA-renamed away under full unroll
        }
    }

    // C layout: VGPR r -> (M = r | r+8, N = lane&15) per 16-lane half
    int mbase = blockIdx.x * 128 + wave * 16 + ((lane >> 4) << 3);
    int col   = lane & 15;
#pragma unroll
    for (int t = 0; t < 8; ++t) {
#pragma unroll
        for (int r = 0; r < 8; ++r) {
            int m = mbase + r;
            if (m < nrows) Y[(long)m * HDIM + t * 16 + col] = acc[t][r];
        }
    }
}

// ---------------- self-loop init: agg = h * dinv^2 ----------------
__global__ void selfloop(const float* __restrict__ h, const float* __restrict__ dinv,
                         float* __restrict__ agg) {
    long i = (long)blockIdx.x * 256 + threadIdx.x;
    if (i < (long)N_NODES * HDIM) {
        float di = dinv[i >> 7];
        agg[i] = h[i] * di * di;
    }
}

// ---------------- edge scatter-add: agg[dst] += h[src] * enorm -----------
// one wave per edge; h/agg (51.2 MB each) are L2-resident (192 MB), so the
// random gather and f32 atomics are served by GL2, not HBM.
__global__ __launch_bounds__(256) void edge_agg(const float* __restrict__ h,
                                                float* __restrict__ agg,
                                                const int* __restrict__ src,
                                                const int* __restrict__ dst,
                                                const float* __restrict__ dinv) {
    int e = blockIdx.x * 8 + (threadIdx.x >> 5);
    if (e >= N_EDGES) return;
    int lane = threadIdx.x & 31;
    int s = src[e], d = dst[e];
    float en = dinv[s] * dinv[d];
    float4 v = *(const float4*)(h + (long)s * HDIM + lane * 4);
    float* p = agg + (long)d * HDIM + lane * 4;
    atomicAdd(p + 0, v.x * en);
    atomicAdd(p + 1, v.y * en);
    atomicAdd(p + 2, v.z * en);
    atomicAdd(p + 3, v.w * en);
}

// ---------------- bias + layernorm + relu (wave per node) ----------------
// Emits the bf16 row for the next layer's GEMM; f32 row only when needed
// (final layer -> pooling).
__global__ __launch_bounds__(256) void ln_relu(const float* __restrict__ agg,
                                               const float* __restrict__ b,
                                               const float* __restrict__ g,
                                               const float* __restrict__ beta,
                                               float* __restrict__ out_f32,
                                               unsigned int* __restrict__ out_bf,
                                               int write_f32) {
    int node = blockIdx.x * 8 + (threadIdx.x >> 5);
    if (node >= N_NODES) return;
    int lane = threadIdx.x & 31;
    float4 v  = *(const float4*)(agg + (long)node * HDIM + lane * 4);
    float4 bb = *(const float4*)(b + lane * 4);
    v.x += bb.x; v.y += bb.y; v.z += bb.z; v.w += bb.w;
    float s  = v.x + v.y + v.z + v.w;
    float s2 = v.x * v.x + v.y * v.y + v.z * v.z + v.w * v.w;
#pragma unroll
    for (int off = 16; off > 0; off >>= 1) {
        s  += __shfl_xor(s,  off, 32);
        s2 += __shfl_xor(s2, off, 32);
    }
    float mu  = s * (1.0f / HDIM);
    float var = s2 * (1.0f / HDIM) - mu * mu;
    float inv = rsqrtf(var + 1e-5f);
    float4 gg = *(const float4*)(g + lane * 4);
    float4 be = *(const float4*)(beta + lane * 4);
    float4 o;
    o.x = fmaxf((v.x - mu) * inv * gg.x + be.x, 0.0f);
    o.y = fmaxf((v.y - mu) * inv * gg.y + be.y, 0.0f);
    o.z = fmaxf((v.z - mu) * inv * gg.z + be.z, 0.0f);
    o.w = fmaxf((v.w - mu) * inv * gg.w + be.w, 0.0f);
    uint2 pk;
    pk.x = pack_bf16(o.x, o.y);
    pk.y = pack_bf16(o.z, o.w);
    *(uint2*)(out_bf + (long)node * (HDIM / 2) + lane * 2) = pk;
    if (write_f32)
        *(float4*)(out_f32 + (long)node * HDIM + lane * 4) = o;
}

// ---------------- mean-pool + linear head ----------------
__global__ void pool_init(float* pooled, float* cnts) {
    int i = blockIdx.x * 256 + threadIdx.x;
    if (i < NGRAPH * HDIM) pooled[i] = 0.0f;
    else if (i < NGRAPH * HDIM + NGRAPH) cnts[i - NGRAPH * HDIM] = 0.0f;
}
__global__ __launch_bounds__(256) void pool_acc(const float* __restrict__ h,
                                                const int* __restrict__ batch,
                                                float* __restrict__ pooled,
                                                float* __restrict__ cnts) {
    int node = blockIdx.x * 8 + (threadIdx.x >> 5);
    if (node >= N_NODES) return;
    int lane = threadIdx.x & 31;
    int gidx = batch[node];
    float4 v = *(const float4*)(h + (long)node * HDIM + lane * 4);
    float* p = pooled + (long)gidx * HDIM + lane * 4;
    atomicAdd(p + 0, v.x);
    atomicAdd(p + 1, v.y);
    atomicAdd(p + 2, v.z);
    atomicAdd(p + 3, v.w);
    if (lane == 0) atomicAdd(&cnts[gidx], 1.0f);
}
__global__ __launch_bounds__(256) void pool_final(const float* __restrict__ pooled,
                                                  const float* __restrict__ cnts,
                                                  const float* __restrict__ Wl,
                                                  const float* __restrict__ bl,
                                                  float* __restrict__ out) {
    int gidx = blockIdx.x * 8 + (threadIdx.x >> 5);
    if (gidx >= NGRAPH) return;
    int lane = threadIdx.x & 31;
    float4 v = *(const float4*)(pooled + (long)gidx * HDIM + lane * 4);
    float4 w = *(const float4*)(Wl + lane * 4);
    float s = v.x * w.x + v.y * w.y + v.z * w.z + v.w * w.w;
#pragma unroll
    for (int off = 16; off > 0; off >>= 1) s += __shfl_xor(s, off, 32);
    if (lane == 0) out[gidx] = s / fmaxf(cnts[gidx], 1.0f) + bl[0];
}

extern "C" void kernel_launch(void* const* d_in, const int* in_sizes, int n_in,
                              void* d_out, int out_size, void* d_ws, size_t ws_size,
                              hipStream_t stream) {
    const float* x     = (const float*)d_in[0];
    const int*   ei    = (const int*)d_in[1];
    const int*   src   = ei;
    const int*   dst   = ei + N_EDGES;
    const int*   batch = (const int*)d_in[2];
    const float* W[3]  = {(const float*)d_in[3],  (const float*)d_in[7],  (const float*)d_in[11]};
    const float* bb[3] = {(const float*)d_in[4],  (const float*)d_in[8],  (const float*)d_in[12]};
    const float* gg[3] = {(const float*)d_in[5],  (const float*)d_in[9],  (const float*)d_in[13]};
    const float* be[3] = {(const float*)d_in[6],  (const float*)d_in[10], (const float*)d_in[14]};
    const float* Wl    = (const float*)d_in[15];
    const float* bl    = (const float*)d_in[16];
    float* out = (float*)d_out;

    // workspace (floats): dinv | hA | hB | pooled | cnts | wsw | xbf
    float* dinv   = (float*)d_ws;
    float* hA     = dinv + ((N_NODES + 255) & ~255);
    float* hB     = hA + (size_t)N_NODES * HDIM;
    float* pooled = hB + (size_t)N_NODES * HDIM;
    float* cnts   = pooled + (size_t)NGRAPH * HDIM;
    unsigned int* wsw = (unsigned int*)(cnts + NGRAPH);
    unsigned int* xbf = wsw + 8192;                    // N x 128 bf16 (64 dw/row)

    deg_init <<<(N_NODES + 255) / 256, 256, 0, stream>>>(dinv);
    deg_acc  <<<(N_EDGES + 255) / 256, 256, 0, stream>>>(dinv, dst);
    deg_rsqrt<<<(N_NODES + 255) / 256, 256, 0, stream>>>(dinv);

    xprep<<<(int)(((long)N_NODES * (HDIM / 2) + 255) / 256), 256, 0, stream>>>(x, xbf);

    float* bufs[2] = {hA, hB};
    float* lastF = nullptr;
    for (int l = 0; l < 3; ++l) {
        float* hG  = bufs[l & 1];        // gemm out (h)
        float* hAg = bufs[(l + 1) & 1];  // aggregation target
        wprep   <<<32, 256, 0, stream>>>(W[l], wsw);
        gemm_xw <<<(N_NODES + 127) / 128, 256, 0, stream>>>(xbf, wsw, hG, N_NODES);
        selfloop<<<(int)(((long)N_NODES * HDIM + 255) / 256), 256, 0, stream>>>(hG, dinv, hAg);
        edge_agg<<<(N_EDGES + 7) / 8, 256, 0, stream>>>(hG, hAg, src, dst, dinv);
        ln_relu <<<(N_NODES + 7) / 8, 256, 0, stream>>>(hAg, bb[l], gg[l], be[l],
                                                        hG, xbf, (l == 2) ? 1 : 0);
        lastF = hG;
    }

    pool_init <<<(NGRAPH * HDIM + NGRAPH + 255) / 256, 256, 0, stream>>>(pooled, cnts);
    pool_acc  <<<(N_NODES + 7) / 8, 256, 0, stream>>>(lastF, batch, pooled, cnts);
    pool_final<<<(NGRAPH + 7) / 8, 256, 0, stream>>>(pooled, cnts, Wl, bl, out);
}
